// PytorchDotProductAttention_35192962023831
// MI455X (gfx1250) — matrix operand order repaired and checked by reading, not compiled, not run on hardware
//
#include <hip/hip_runtime.h>
#include <hip/hip_bf16.h>

// ---------------------------------------------------------------------------
// Problem sizes (fixed by the reference)
// ---------------------------------------------------------------------------
#define EMB   1024
#define SEQ   2048
#define BATCH 4
#define ROWS  (SEQ * BATCH)        // 8192 flattened (s,b) rows
#define QKV_N (3 * EMB)            // 3072
#define KSTEP 32                   // K per WMMA stage
#define LDSW  40                   // padded LDS row stride (bf16 elems) ->
                                   // 20-dword stride, conflict-free frag reads

typedef __bf16 bf16_t;
typedef __attribute__((ext_vector_type(16))) __bf16 v16bf;
typedef __attribute__((ext_vector_type(8)))  __bf16 v8bf;
typedef __attribute__((ext_vector_type(8)))  float  v8f;
typedef __attribute__((ext_vector_type(4)))  int    v4i;

#if __has_builtin(__builtin_amdgcn_global_load_async_to_lds_b128)
#define HAVE_ASYNC_LDS 1
#else
#define HAVE_ASYNC_LDS 0
#endif

// ---------------------------------------------------------------------------
// CDNA5 wait helpers (ASYNCcnt / DScnt)
// ---------------------------------------------------------------------------
template <int N>
__device__ __forceinline__ void wait_asynccnt() {
#if __has_builtin(__builtin_amdgcn_s_wait_asynccnt)
  __builtin_amdgcn_s_wait_asynccnt(N);
#else
  asm volatile("s_wait_asynccnt %0" ::"i"(N) : "memory");
#endif
}

__device__ __forceinline__ void wait_dscnt0() {
#if __has_builtin(__builtin_amdgcn_s_wait_dscnt)
  __builtin_amdgcn_s_wait_dscnt(0);
#else
  asm volatile("s_wait_dscnt 0x0" ::: "memory");
#endif
}

#if HAVE_ASYNC_LDS
// One 16-byte async global->LDS copy (per-lane addresses, ASYNCcnt-tracked).
// Builtin signature (per hipcc diagnostic): (v4i AS(1)*, v4i AS(3)*, Ii, Ii).
__device__ __forceinline__ void async_cp16(const bf16_t* g, bf16_t* l) {
  __builtin_amdgcn_global_load_async_to_lds_b128(
      (__attribute__((address_space(1))) v4i*)(v4i*)g,
      (__attribute__((address_space(3))) v4i*)(v4i*)l, 0, 0);
}
#endif

// ---------------------------------------------------------------------------
// f32 -> bf16 conversion (4 elements / thread, fully coalesced)
// ---------------------------------------------------------------------------
__global__ void cvt_f32_to_bf16(const float* __restrict__ src,
                                bf16_t* __restrict__ dst, int n4) {
  int i = blockIdx.x * blockDim.x + threadIdx.x;
  if (i >= n4) return;
  const float4 f = ((const float4*)src)[i];
  bf16_t o[4] = {(bf16_t)f.x, (bf16_t)f.y, (bf16_t)f.z, (bf16_t)f.w};
  ((ulong1*)dst)[i] = *(const ulong1*)o;   // 8-byte store of 4 bf16
}

// ---------------------------------------------------------------------------
// Block-cooperative WMMA mainloop.
//   C_block = A[64 rows, K] x Bt[128 rows, K]^T   (both row-major, K-contig)
// 256 threads = 8 waves arranged 4(M) x 2(N); each wave owns 16x64 (4 accs).
// K-step tiles (A 64x32, B 128x32 bf16) are double-buffered in LDS and
// filled with GLOBAL_LOAD_ASYNC_TO_LDS_B128 (3 x 16B per thread per stage).
// ---------------------------------------------------------------------------
__device__ __forceinline__ void gemm_block_tile(const bf16_t* __restrict__ A,
                                                const bf16_t* __restrict__ Bt,
                                                int lda, int ldb, int K,
                                                v8f acc[4]) {
  __shared__ alignas(16) bf16_t As[2][64 * LDSW];
  __shared__ alignas(16) bf16_t Bs[2][128 * LDSW];

  const int tid  = threadIdx.x;
  const int lane = tid & 31;
  const int wave = tid >> 5;
  const int lmod = lane & 15, lhalf = lane >> 4;
  const int m0w  = (wave >> 1) * 16;   // wave row offset inside block tile
  const int n0w  = (wave & 1) * 64;    // wave col offset inside block tile

  // cooperative-copy coordinates: 16B chunks, 4 chunks per 32-elem row
  const int arow  = tid >> 2;                 // A: 64 rows x 4 chunks
  const int brow0 = tid >> 2;                 // B: 128 rows x 4 chunks,
  const int brow1 = (tid + 256) >> 2;         //    2 chunks per thread
  const int ccol  = (tid & 3) * 8;            // bf16 elems (16B)

  const bf16_t* gA  = A  + ((long)blockIdx.y * 64  + arow)  * lda + ccol;
  const bf16_t* gB0 = Bt + ((long)blockIdx.x * 128 + brow0) * ldb + ccol;
  const bf16_t* gB1 = Bt + ((long)blockIdx.x * 128 + brow1) * ldb + ccol;
  const int la_off  = arow  * LDSW + ccol;
  const int lb0_off = brow0 * LDSW + ccol;
  const int lb1_off = brow1 * LDSW + ccol;

  const int nst = K / KSTEP;

  union ABf { v16bf v; struct { v8bf lo, hi; } p; };

#if HAVE_ASYNC_LDS
  // ---- async double-buffered pipeline ----
  {
    // prologue: stages 0 and 1 in flight
    async_cp16(gA, &As[0][la_off]);
    async_cp16(gB0, &Bs[0][lb0_off]);
    async_cp16(gB1, &Bs[0][lb1_off]);
    if (nst > 1) {
      async_cp16(gA + KSTEP, &As[1][la_off]);
      async_cp16(gB0 + KSTEP, &Bs[1][lb0_off]);
      async_cp16(gB1 + KSTEP, &Bs[1][lb1_off]);
    }
  }
  for (int i = 0; i < nst; ++i) {
    const int buf = i & 1;
    if (i + 1 < nst) wait_asynccnt<3>();   // stage i landed, i+1 in flight
    else             wait_asynccnt<0>();
    __syncthreads();

    // fragment reads from LDS (ISA wave32 layouts)
    ABf a, b[4];
    const bf16_t* la = &As[buf][(m0w + lmod) * LDSW + 8 * lhalf];
    a.p.lo = *(const v8bf*)(la);
    a.p.hi = *(const v8bf*)(la + 16);
#pragma unroll
    for (int j = 0; j < 4; ++j) {
      const bf16_t* lb = &Bs[buf][(n0w + j * 16 + lmod) * LDSW + 8 * lhalf];
      b[j].p.lo = *(const v8bf*)(lb);
      b[j].p.hi = *(const v8bf*)(lb + 16);
    }
    wait_dscnt0();        // fragments in VGPRs before buffer is overwritten
    __syncthreads();      // ...by every wave

    if (i + 2 < nst) {    // refill this buffer with stage i+2
      const long k = (long)(i + 2) * KSTEP;
      async_cp16(gA + k, &As[buf][la_off]);
      async_cp16(gB0 + k, &Bs[buf][lb0_off]);
      async_cp16(gB1 + k, &Bs[buf][lb1_off]);
    }

#pragma unroll
    for (int j = 0; j < 4; ++j)
      acc[j] = __builtin_amdgcn_wmma_f32_16x16x32_bf16(
          false, a.v, false, b[j].v, (short)0, acc[j], false, false);
  }
#else
  // ---- fallback: synchronous global->LDS staging, single buffer ----
  for (int i = 0; i < nst; ++i) {
    const long k = (long)i * KSTEP;
    __syncthreads();
    *(v8bf*)&As[0][la_off]  = *(const v8bf*)(gA + k);
    *(v8bf*)&Bs[0][lb0_off] = *(const v8bf*)(gB0 + k);
    *(v8bf*)&Bs[0][lb1_off] = *(const v8bf*)(gB1 + k);
    __syncthreads();

    ABf a, b[4];
    const bf16_t* la = &As[0][(m0w + lmod) * LDSW + 8 * lhalf];
    a.p.lo = *(const v8bf*)(la);
    a.p.hi = *(const v8bf*)(la + 16);
#pragma unroll
    for (int j = 0; j < 4; ++j) {
      const bf16_t* lb = &Bs[0][(n0w + j * 16 + lmod) * LDSW + 8 * lhalf];
      b[j].p.lo = *(const v8bf*)(lb);
      b[j].p.hi = *(const v8bf*)(lb + 16);
    }
#pragma unroll
    for (int j = 0; j < 4; ++j)
      acc[j] = __builtin_amdgcn_wmma_f32_16x16x32_bf16(
          false, a.v, false, b[j].v, (short)0, acc[j], false, false);
  }
#endif
}

// Wave/output coordinates (match gemm_block_tile's intra-block layout).
__device__ __forceinline__ void tile_coords(int& m0, int& n0, int& lane) {
  lane = threadIdx.x & 31;
  const int wave = threadIdx.x >> 5;
  m0 = blockIdx.y * 64 + (wave >> 1) * 16;
  n0 = blockIdx.x * 128 + (wave & 1) * 64;
}

// ---------------------------------------------------------------------------
// GEMM 1: qkv = Qbf[ROWS,EMB] x Winbf[QKV_N,EMB]^T
// Epilogue: split into q (scaled, [B,S,E] bf16), k ([B,S,E] bf16),
//           v transposed ([B,E,S] bf16) so weights@v is K-contiguous later.
// ---------------------------------------------------------------------------
__global__ void gemm_qkv(const bf16_t* __restrict__ Qbf,
                         const bf16_t* __restrict__ Winbf,
                         bf16_t* __restrict__ qbf,
                         bf16_t* __restrict__ kbf,
                         bf16_t* __restrict__ vT) {
  v8f acc[4] = {};
  gemm_block_tile(Qbf, Winbf, EMB, EMB, EMB, acc);

  int m0, n0, lane;
  tile_coords(m0, n0, lane);
  const float scale = 0.03125f;  // 1/sqrt(1024)
  const int lmod = lane & 15, lhalf = lane >> 4;
#pragma unroll
  for (int j = 0; j < 4; ++j) {
    const int n = n0 + j * 16 + lmod;
#pragma unroll
    for (int r = 0; r < 8; ++r) {
      const int m = m0 + r + 8 * lhalf;     // m = s*BATCH + b
      const int s = m >> 2, b = m & 3;      // BATCH == 4
      const float val = acc[j][r];
      if (n < EMB) {
        qbf[((long)b * SEQ + s) * EMB + n] = (bf16_t)(val * scale);
      } else if (n < 2 * EMB) {
        kbf[((long)b * SEQ + s) * EMB + (n - EMB)] = (bf16_t)val;
      } else {
        vT[((long)b * EMB + (n - 2 * EMB)) * SEQ + s] = (bf16_t)val;
      }
    }
  }
}

// ---------------------------------------------------------------------------
// GEMM 2 (per batch, blockIdx.z): scores[T,S] = q[T,E] x k[S,E]^T  (f32 out)
// Written straight into the `weights` region of d_out (softmaxed in place).
// ---------------------------------------------------------------------------
__global__ void gemm_scores(const bf16_t* __restrict__ qbf,
                            const bf16_t* __restrict__ kbf,
                            float* __restrict__ scores) {
  const int bz = blockIdx.z;
  v8f acc[4] = {};
  gemm_block_tile(qbf + (long)bz * SEQ * EMB, kbf + (long)bz * SEQ * EMB,
                  EMB, EMB, EMB, acc);

  float* C = scores + (long)bz * SEQ * SEQ;
  int m0, n0, lane;
  tile_coords(m0, n0, lane);
  const int lmod = lane & 15, lhalf = lane >> 4;
#pragma unroll
  for (int j = 0; j < 4; ++j) {
    const int n = n0 + j * 16 + lmod;
#pragma unroll
    for (int r = 0; r < 8; ++r)
      C[(long)(m0 + r + 8 * lhalf) * SEQ + n] = acc[j][r];
  }
}

// ---------------------------------------------------------------------------
// Row softmax over 2048-wide rows, in place (f32) + bf16 copy for next GEMM.
// One 256-thread block per row; 8 register-resident elems/thread.
// ---------------------------------------------------------------------------
__global__ void softmax_rows(float* __restrict__ w, bf16_t* __restrict__ wbf) {
  const int tid = threadIdx.x;
  const long row = blockIdx.x;
  float* p = w + row * (long)SEQ;

  float v[8];
  float m = -INFINITY;
#pragma unroll
  for (int i = 0; i < 8; ++i) {
    v[i] = p[tid + i * 256];
    m = fmaxf(m, v[i]);
  }
  __shared__ float red[256];
  red[tid] = m; __syncthreads();
  for (int s = 128; s > 0; s >>= 1) {
    if (tid < s) red[tid] = fmaxf(red[tid], red[tid + s]);
    __syncthreads();
  }
  m = red[0]; __syncthreads();

  float sum = 0.f;
#pragma unroll
  for (int i = 0; i < 8; ++i) { v[i] = __expf(v[i] - m); sum += v[i]; }
  red[tid] = sum; __syncthreads();
  for (int s = 128; s > 0; s >>= 1) {
    if (tid < s) red[tid] += red[tid + s];
    __syncthreads();
  }
  const float inv = 1.0f / red[0];

  bf16_t* pb = wbf + row * (long)SEQ;
#pragma unroll
  for (int i = 0; i < 8; ++i) {
    const float o = v[i] * inv;
    p[tid + i * 256]  = o;
    pb[tid + i * 256] = (bf16_t)o;
  }
}

// ---------------------------------------------------------------------------
// GEMM 3 (per batch): attnout[T,E] = weights[T,S] x vT[E,S]^T   (bf16 out)
// ---------------------------------------------------------------------------
__global__ void gemm_attnout(const bf16_t* __restrict__ wbf,
                             const bf16_t* __restrict__ vT,
                             bf16_t* __restrict__ aobf) {
  const int bz = blockIdx.z;
  v8f acc[4] = {};
  gemm_block_tile(wbf + (long)bz * SEQ * SEQ, vT + (long)bz * EMB * SEQ,
                  SEQ, SEQ, SEQ, acc);

  bf16_t* C = aobf + (long)bz * SEQ * EMB;
  int m0, n0, lane;
  tile_coords(m0, n0, lane);
  const int lmod = lane & 15, lhalf = lane >> 4;
#pragma unroll
  for (int j = 0; j < 4; ++j) {
    const int n = n0 + j * 16 + lmod;
#pragma unroll
    for (int r = 0; r < 8; ++r)
      C[(long)(m0 + r + 8 * lhalf) * EMB + n] = (bf16_t)acc[j][r];
  }
}

// ---------------------------------------------------------------------------
// GEMM 4: out = attnout[B*T,E] x Wout[E,E]^T, permuted to [T,B,E] f32.
// ---------------------------------------------------------------------------
__global__ void gemm_outproj(const bf16_t* __restrict__ aobf,
                             const bf16_t* __restrict__ Woutbf,
                             float* __restrict__ out) {
  v8f acc[4] = {};
  gemm_block_tile(aobf, Woutbf, EMB, EMB, EMB, acc);

  int m0, n0, lane;
  tile_coords(m0, n0, lane);
  const int lmod = lane & 15, lhalf = lane >> 4;
#pragma unroll
  for (int j = 0; j < 4; ++j) {
    const int n = n0 + j * 16 + lmod;
#pragma unroll
    for (int r = 0; r < 8; ++r) {
      const int m = m0 + r + 8 * lhalf;    // m = b*SEQ + t
      const int b = m / SEQ, t = m % SEQ;
      out[((long)t * BATCH + b) * EMB + n] = acc[j][r];
    }
  }
}

// ---------------------------------------------------------------------------
// Host-side launch
// ---------------------------------------------------------------------------
extern "C" void kernel_launch(void* const* d_in, const int* in_sizes, int n_in,
                              void* d_out, int out_size, void* d_ws, size_t ws_size,
                              hipStream_t stream) {
  const float* Q     = (const float*)d_in[0];
  // d_in[1] (K) and d_in[2] (V) are ignored by the reference.
  const float* W_in  = (const float*)d_in[3];
  const float* W_out = (const float*)d_in[4];

  float* out_att = (float*)d_out;                          // [T,B,E]   8.4M f32
  float* out_w   = (float*)d_out + (long)ROWS * EMB;       // [B,T,S]  16.8M f32

  // Workspace layout (bytes)
  char* ws = (char*)d_ws;
  bf16_t* qbf    = (bf16_t*)(ws);                           // 16 MB [B,S,E]
  bf16_t* kbf    = (bf16_t*)(ws + (1L << 24));              // 16 MB [B,S,E]
  bf16_t* vT     = (bf16_t*)(ws + 2 * (1L << 24));          // 16 MB [B,E,S]
  bf16_t* wbf    = (bf16_t*)(ws + 3 * (1L << 24));          // 32 MB [B,T,S]
  bf16_t* aobf   = (bf16_t*)(ws + 5 * (1L << 24));          // 16 MB [B,T,E]
  bf16_t* Qbf    = (bf16_t*)(ws + 6 * (1L << 24));          // 16 MB [ROWS,E]
  bf16_t* Winbf  = (bf16_t*)(ws + 7 * (1L << 24));          //  6 MB [3E,E]
  bf16_t* Woutbf = (bf16_t*)(ws + 7 * (1L << 24) + 6291456);//  2 MB [E,E]

  const int TB = 256;

  // 0) precision conversion
  cvt_f32_to_bf16<<<(ROWS * EMB / 4 + TB - 1) / TB, TB, 0, stream>>>(Q, Qbf, ROWS * EMB / 4);
  cvt_f32_to_bf16<<<(QKV_N * EMB / 4 + TB - 1) / TB, TB, 0, stream>>>(W_in, Winbf, QKV_N * EMB / 4);
  cvt_f32_to_bf16<<<(EMB * EMB / 4 + TB - 1) / TB, TB, 0, stream>>>(W_out, Woutbf, EMB * EMB / 4);

  // 1) QKV projection (+ split/scale/transpose epilogue)
  gemm_qkv<<<dim3(QKV_N / 128, ROWS / 64), TB, 0, stream>>>(Qbf, Winbf, qbf, kbf, vT);

  // 2) attention scores -> d_out weights region (f32)
  gemm_scores<<<dim3(SEQ / 128, SEQ / 64, BATCH), TB, 0, stream>>>(qbf, kbf, out_w);

  // 3) softmax in place (+ bf16 copy)
  softmax_rows<<<BATCH * SEQ, TB, 0, stream>>>(out_w, wbf);

  // 4) weights @ v
  gemm_attnout<<<dim3(EMB / 128, SEQ / 64, BATCH), TB, 0, stream>>>(wbf, vT, aobf);

  // 5) output projection, permuted to [T,B,E]
  gemm_outproj<<<dim3(EMB / 128, ROWS / 64), TB, 0, stream>>>(aobf, Woutbf, out_att);
}
